// CoxSGDLossFn_62105227100318
// MI455X (gfx1250) — compile-verified, matrix operand order checked by reading, not run on hardware
//
#include <hip/hip_runtime.h>

// CoxSGD ranking loss, MI455X (gfx1250, wave32).
// Algorithm: per (task,row) top-2 selection over the JAX threefry random
// matrix (1 wave per row, columns staged in LDS via async-to-LDS), then
// sparse scatter for column sums and WMMA-fp32 final reductions.

#define N 4096
#define T 4
#define NT (N * T)
#define REG_W 0.05f

typedef __attribute__((ext_vector_type(2)))  float    v2f;
typedef __attribute__((ext_vector_type(8)))  float    v8f;
typedef __attribute__((ext_vector_type(16))) _Float16 v16h;

#if __has_builtin(__builtin_amdgcn_global_load_async_to_lds_b32)
#define HAVE_ASYNC_LDS 1
// Prototype (from hipcc diagnostics): (as1 int*, as3 int*, imm int, imm int)
typedef __attribute__((address_space(1))) int* as1_i32p;
typedef __attribute__((address_space(3))) int* as3_i32p;
#endif

__device__ __forceinline__ unsigned rotl32(unsigned x, unsigned r) {
  return (x << r) | (x >> (32u - r));
}

// JAX threefry2x32 stream for jax.random.uniform(key(42), (T,N,N), f32).
// key = (0,42); counters split in halves: bits[i] = i<2^25 ? out0(i,i+2^25)
// : out1(i-2^25, i). uniform = bitcast((bits>>9)|0x3f800000) - 1.
__device__ __forceinline__ float jax_uniform_u42(unsigned idx) {
  const unsigned k0 = 0u, k1 = 42u;
  const unsigned ks2 = k0 ^ k1 ^ 0x1BD11BDAu;
  const unsigned half = 1u << 25;  // total count = 2^26
  const bool lo = idx < half;
  unsigned x0 = lo ? idx : idx - half;
  unsigned x1 = lo ? idx + half : idx;
  x0 += k0; x1 += k1;
#define TF_R(r) { x0 += x1; x1 = rotl32(x1, r); x1 ^= x0; }
  TF_R(13u) TF_R(15u) TF_R(26u) TF_R(6u)
  x0 += k1;  x1 += ks2 + 1u;
  TF_R(17u) TF_R(29u) TF_R(16u) TF_R(24u)
  x0 += ks2; x1 += k0 + 2u;
  TF_R(13u) TF_R(15u) TF_R(26u) TF_R(6u)
  x0 += k0;  x1 += k1 + 3u;
  TF_R(17u) TF_R(29u) TF_R(16u) TF_R(24u)
  x0 += k1;  x1 += ks2 + 4u;
  TF_R(13u) TF_R(15u) TF_R(26u) TF_R(6u)
  x0 += ks2; x1 += k0 + 5u;
#undef TF_R
  unsigned bits = lo ? x0 : x1;
  return __uint_as_float((bits >> 9) | 0x3f800000u) - 1.0f;
}

// ---------------------------------------------------------------------------
// Wave-level exact-fp32 sum of 2048 values via V_WMMA_F32_16X16X4_F32.
// B = ones(4x16) => D[m,n] = sum_k A[m,k]; accumulate C across chunks; every
// D row is replicated 16x across columns, so sum(all 256 lanes*vgprs)/16 ==
// total. Fixed order => deterministic.
// ---------------------------------------------------------------------------
template <typename F>
__device__ __forceinline__ float wave_sum_2048(int lane, unsigned base, F getv) {
#if __has_builtin(__builtin_amdgcn_wmma_f32_16x16x4_f32)
  v2f bone; bone[0] = 1.0f; bone[1] = 1.0f;
  v8f acc = {0.f, 0.f, 0.f, 0.f, 0.f, 0.f, 0.f, 0.f};
  for (unsigned k = 0u; k < 2048u; k += 64u) {
    v2f a;
    a[0] = getv(base + k + 2u * (unsigned)lane);
    a[1] = getv(base + k + 2u * (unsigned)lane + 1u);
    acc = __builtin_amdgcn_wmma_f32_16x16x4_f32(false, a, false, bone,
                                                (short)0, acc, false, false);
  }
  float s = acc[0] + acc[1] + acc[2] + acc[3] + acc[4] + acc[5] + acc[6] + acc[7];
  for (int off = 16; off > 0; off >>= 1) s += __shfl_xor(s, off, 32);
  return s * 0.0625f;
#elif __has_builtin(__builtin_amdgcn_wmma_f32_16x16x32_f16)
  v16h bone;
#pragma unroll
  for (int e = 0; e < 16; ++e) bone[e] = (_Float16)1.0f;
  v8f acc = {0.f, 0.f, 0.f, 0.f, 0.f, 0.f, 0.f, 0.f};
  for (unsigned k = 0u; k < 2048u; k += 512u) {
    v16h a;
#pragma unroll
    for (int e = 0; e < 16; ++e)
      a[e] = (_Float16)getv(base + k + 16u * (unsigned)lane + (unsigned)e);
    acc = __builtin_amdgcn_wmma_f32_16x16x32_f16(false, a, false, bone,
                                                 (short)0, acc, false, false);
  }
  float s = acc[0] + acc[1] + acc[2] + acc[3] + acc[4] + acc[5] + acc[6] + acc[7];
  for (int off = 16; off > 0; off >>= 1) s += __shfl_xor(s, off, 32);
  return s * 0.0625f;
#else
  float s = 0.f;
  for (unsigned k = (unsigned)lane; k < 2048u; k += 32u) s += getv(base + k);
  for (int off = 16; off > 0; off >>= 1) s += __shfl_xor(s, off, 32);
  return s;
#endif
}

// ---------------------------------------------------------------------------
__global__ __launch_bounds__(256) void cox_zero(unsigned* __restrict__ colsum) {
  int i = blockIdx.x * 256 + threadIdx.x;
  if (i < NT) colsum[i] = 0u;
}

__global__ __launch_bounds__(256) void cox_maxp(const float* __restrict__ y_pred,
                                                float* __restrict__ maxp) {
  __shared__ float red[256];
  const int t = blockIdx.x, tid = threadIdx.x;
  float m = -3.402823466e38f;
  for (int i = tid; i < N; i += 256) m = fmaxf(m, y_pred[i * T + t]);
  red[tid] = m;
  __syncthreads();
  for (int s = 128; s > 0; s >>= 1) {
    if (tid < s) red[tid] = fmaxf(red[tid], red[tid + s]);
    __syncthreads();
  }
  if (tid == 0) maxp[t] = red[0];
}

// One wave32 per (task, row): top-2 eligible random draws -> surviving pair
// entries; writes row_loss and scatters integer column-sum increments.
__global__ __launch_bounds__(256) void cox_main(const float* __restrict__ y_pred,
                                                const float* __restrict__ length,
                                                const float* __restrict__ event,
                                                const float* __restrict__ maxp_ws,
                                                float* __restrict__ rowloss,
                                                unsigned* __restrict__ colsum) {
  __shared__ float s_len[N];
  __shared__ float s_pred[N];
  const int t = blockIdx.y;
  const int tid = threadIdx.x;

  // Stage this task's length/pred columns (strided 16B in global) into LDS
  // with the CDNA5 async-to-LDS path.
  for (int j = tid; j < N; j += 256) {
#ifdef HAVE_ASYNC_LDS
    __builtin_amdgcn_global_load_async_to_lds_b32(
        (as1_i32p)(uintptr_t)(length + (size_t)j * T + t),
        (as3_i32p)(unsigned)(uintptr_t)(&s_len[j]),
        0, 0);
    __builtin_amdgcn_global_load_async_to_lds_b32(
        (as1_i32p)(uintptr_t)(y_pred + (size_t)j * T + t),
        (as3_i32p)(unsigned)(uintptr_t)(&s_pred[j]),
        0, 0);
#else
    s_len[j]  = length[j * T + t];
    s_pred[j] = y_pred[j * T + t];
#endif
  }
#ifdef HAVE_ASYNC_LDS
#if __has_builtin(__builtin_amdgcn_s_wait_asynccnt)
  __builtin_amdgcn_s_wait_asynccnt(0);
#else
  asm volatile("s_wait_asynccnt 0" ::: "memory");
#endif
#endif
  __syncthreads();

  const int wave = tid >> 5, lane = tid & 31;
  const int i = blockIdx.x * 8 + wave;
  const float ln_i = s_len[i];
  const float pred_i = s_pred[i];
  const float ev_i = event[i * T + t];
  const float maxp = maxp_ws[t];

  // Lane-local top-2 of r[i,j] over eligible j, then wave32 butterfly merge.
  float v1 = -1.f, v2 = -1.f;
  int i1 = -1, i2 = -1;
  if (ev_i > 0.5f) {
    for (int j = lane; j < N; j += 32) {
      const float lj = s_len[j];
      const unsigned idx = (unsigned)t * (unsigned)(N * N) + (unsigned)i * N + (unsigned)j;
      const float u = jax_uniform_u42(idx);
      const float v = (lj > ln_i) ? u : -1.f;
      if (v > v1) { v2 = v1; i2 = i1; v1 = v; i1 = j; }
      else if (v > v2) { v2 = v; i2 = j; }
    }
  }
  for (int off = 16; off > 0; off >>= 1) {
    const float b1 = __shfl_xor(v1, off, 32);
    const int  bi1 = __shfl_xor(i1, off, 32);
    const float b2 = __shfl_xor(v2, off, 32);
    const int  bi2 = __shfl_xor(i2, off, 32);
    if (b1 > v1) {
      if (v1 >= b2) { v2 = v1; i2 = i1; } else { v2 = b2; i2 = bi2; }
      v1 = b1; i1 = bi1;
    } else if (b1 > v2) { v2 = b1; i2 = bi1; }
  }

  if (lane == 0) {
    const int count = (v1 >= 0.f ? 1 : 0) + (v2 >= 0.f ? 1 : 0);
    const bool valid = count > 0;
    const float row_max = maxp - pred_i;  // sd.max(axis=1) = maxp - pred_i
    float lt = 0.f;
    if (valid) lt += expf(pred_i - maxp);          // diagonal term
    if (count >= 1) lt += expf(s_pred[i1] - maxp); // exp(sd - row_max)
    if (count >= 2) lt += expf(s_pred[i2] - maxp);
    rowloss[t * N + i] = valid ? (row_max + logf(lt)) : 0.f;
    if (count >= 1) atomicAdd(&colsum[t * N + i1], 1u);
    if (count >= 2) atomicAdd(&colsum[t * N + i2], 1u);
    if (valid) atomicAdd(&colsum[t * N + i], 1u);
  }
}

// Final deterministic reduction: 8 waves, each WMMA-sums a 2048-chunk of the
// row losses and of the |colsum * pred| regularizer terms.
__global__ __launch_bounds__(256) void cox_final(const float* __restrict__ y_pred,
                                                 const float* __restrict__ rowloss,
                                                 const unsigned* __restrict__ colsum,
                                                 float* __restrict__ out) {
  __shared__ float p_loss[8];
  __shared__ float p_reg[8];
  const int tid = threadIdx.x, wave = tid >> 5, lane = tid & 31;
  const unsigned base = (unsigned)wave * 2048u;

  const float sl = wave_sum_2048(lane, base,
      [=](unsigned k) -> float { return rowloss[k]; });
  const float sr = wave_sum_2048(lane, base,
      [=](unsigned k) -> float {
        const unsigned tt = k >> 12, jj = k & 4095u;
        return fabsf((float)colsum[k] * y_pred[jj * T + tt]);
      });

  if (lane == 0) { p_loss[wave] = sl; p_reg[wave] = sr; }
  __syncthreads();
  if (tid == 0) {
    float L = 0.f, R = 0.f;
    for (int w = 0; w < 8; ++w) { L += p_loss[w]; R += p_reg[w]; }
    out[0] = L + REG_W * R;
  }
}

// ---------------------------------------------------------------------------
extern "C" void kernel_launch(void* const* d_in, const int* in_sizes, int n_in,
                              void* d_out, int out_size, void* d_ws, size_t ws_size,
                              hipStream_t stream) {
  (void)in_sizes; (void)n_in; (void)out_size; (void)ws_size;
  const float* y_pred = (const float*)d_in[0];
  const float* length = (const float*)d_in[1];
  const float* event  = (const float*)d_in[2];
  float* out = (float*)d_out;

  // Workspace layout: [0,256): maxp[4]; [256, 256+64K): colsum (u32, T*N);
  // [256+64K, 256+128K): rowloss (f32, T*N).
  float*    maxp    = (float*)d_ws;
  unsigned* colsum  = (unsigned*)((char*)d_ws + 256);
  float*    rowloss = (float*)((char*)d_ws + 256 + (size_t)NT * 4);

  cox_zero<<<dim3((NT + 255) / 256), dim3(256), 0, stream>>>(colsum);
  cox_maxp<<<dim3(T), dim3(256), 0, stream>>>(y_pred, maxp);
  cox_main<<<dim3(N / 8, T), dim3(256), 0, stream>>>(y_pred, length, event,
                                                     maxp, rowloss, colsum);
  cox_final<<<dim3(1), dim3(256), 0, stream>>>(y_pred, rowloss, colsum, out);
}
